// ChannelWiseFC2d_82626580841131
// MI455X (gfx1250) — matrix operand order repaired and checked by reading, not compiled, not run on hardware
//
#include <hip/hip_runtime.h>

// CDNA5 / gfx1250, wave32.
// Pipeline:
//   1) bitonic_sort_rows: per-(n,c) descending sort of 1024 fp32; emits packed
//      bf16 hi/lo planes into d_ws (needs 2 * 16384 * 512 * 4 = 64 MiB + pad).
//   2) grouped_gemm_bf16x3: per-channel GEMM on the bf16 XDL path using the
//      hi/lo split (hi*hi + hi*lo + lo*hi) -> ~fp32 accuracy at bf16 rate.
//      Bias + sigmoid epilogue in fp32.

typedef __attribute__((ext_vector_type(16))) __bf16 v16bf;
typedef __attribute__((ext_vector_type(8)))  float  v8f;

#define N_  256
#define C_  64
#define KF  1024   // in_features
#define OF  1024   // out_features
#define KD  512    // packed dwords per row (KF/2)

__device__ __forceinline__ unsigned int bf16h_rne(float f) {
  // round-to-nearest-even fp32 -> bf16 (bits in [15:0])
  const unsigned int u = __float_as_uint(f);
  return (u + 0x7FFFu + ((u >> 16) & 1u)) >> 16;
}
__device__ __forceinline__ float bf16_to_f32(unsigned int h) {
  return __uint_as_float(h << 16);
}

// ---------------------------------------------------------------------------
// Kernel 1: bitonic sort (descending) of each length-1024 row, then emit
// packed bf16 hi/lo planes (hi = rne(x), lo = rne(x - hi)).
// ---------------------------------------------------------------------------
__global__ __launch_bounds__(256) void bitonic_sort_rows(
    const float* __restrict__ x,
    unsigned int* __restrict__ zh, unsigned int* __restrict__ zl) {
  __shared__ float s[KF];
  const int row = blockIdx.x;                 // 0 .. N*C-1
  const float* src = x + (size_t)row * KF;
  const int t = threadIdx.x;

  #pragma unroll
  for (int i = t; i < KF; i += 256) s[i] = src[i];
  __syncthreads();

  for (int k = 2; k <= KF; k <<= 1) {
    for (int j = k >> 1; j > 0; j >>= 1) {
      #pragma unroll
      for (int i = t; i < KF; i += 256) {
        const int ixj = i ^ j;
        if (ixj > i) {
          const float a = s[i];
          const float b = s[ixj];
          const bool desc = ((i & k) == 0);   // -> overall descending
          if (desc ? (a < b) : (a > b)) { s[i] = b; s[ixj] = a; }
        }
      }
      __syncthreads();
    }
  }

  #pragma unroll
  for (int d = t; d < KD; d += 256) {
    const float a = s[2 * d];
    const float b = s[2 * d + 1];
    const unsigned int ha = bf16h_rne(a);
    const unsigned int hb = bf16h_rne(b);
    const unsigned int la = bf16h_rne(a - bf16_to_f32(ha));
    const unsigned int lb = bf16h_rne(b - bf16_to_f32(hb));
    zh[(size_t)row * KD + d] = ha | (hb << 16);
    zl[(size_t)row * KD + d] = la | (lb << 16);
  }
}

// ---------------------------------------------------------------------------
// Kernel 2: per-channel GEMM via V_WMMA_F32_16X16X32_BF16, bf16x3 split.
// Block: 1 channel x 128 N-rows x 64 O-cols, 256 threads (8 waves).
// Wave: 2 col-tiles x 2 row-tiles of 16x16 (4 v8f accumulators).
// LDS: 4 planes (Zh/Zl/Wh/Wl), packed bf16 pairs, padded stride 36 dwords.
// ---------------------------------------------------------------------------
union Frag {
  v16bf bf;
  struct { uint4 lo; uint4 hi; } q;
};

__device__ __forceinline__ v8f wmma_bf16(v16bf a, v16bf b, v8f c) {
  return __builtin_amdgcn_wmma_f32_16x16x32_bf16(
      false, a, false, b, (short)0, c, false, false);
}

__global__ __launch_bounds__(256) void grouped_gemm_bf16x3(
    const unsigned int* __restrict__ zh, const unsigned int* __restrict__ zl,
    const float* __restrict__ Wt, const float* __restrict__ bias,
    float* __restrict__ out) {
  constexpr int KB  = 64;   // K per LDS stage (floats) = 32 packed dwords
  constexpr int LDW = 36;   // LDS dword stride per row (32 data + 4 pad)
  __shared__ unsigned int Zh[128 * LDW];
  __shared__ unsigned int Zl[128 * LDW];
  __shared__ unsigned int Wh[64 * LDW];
  __shared__ unsigned int Wl[64 * LDW];

  const int c    = blockIdx.z;
  const int row0 = blockIdx.y * 128;     // N base
  const int col0 = blockIdx.x * 64;      // O base

  const int t    = threadIdx.x;
  const int lane = t & 31;
  const int wv   = t >> 5;               // 0..7
  const int half = lane >> 4;            // 0: lanes 0-15, 1: lanes 16-31
  const int lm   = lane & 15;
  const int cp   = wv & 1;               // col pair: tiles cp*2+j (j=0,1)
  const int rh   = wv >> 1;              // row group: rows rh*32 + r*16

  const float*        Wc  = Wt + ((size_t)c * OF + col0) * KF;
  const unsigned int* zhb = zh + (size_t)row0 * KD;
  const unsigned int* zlb = zl + (size_t)row0 * KD;

  v8f acc[4] = {};   // acc[j*2 + r]

  for (int kb = 0; kb < KF; kb += KB) {
    const int kd = kb >> 1;   // packed dword base in global Z rows

    // --- stage Z (pure copy of pre-packed bf16 planes) --------------------
    for (int idx = t; idx < 1024; idx += 256) {
      const int r  = idx >> 3;            // 0..127
      const int ch = (idx & 7) << 2;      // dword offset 0..28
      const uint4 h4 = *(const uint4*)(zhb + (size_t)r * KD + kd + ch);
      *(uint4*)(&Zh[r * LDW + ch]) = h4;
      const uint4 l4 = *(const uint4*)(zlb + (size_t)r * KD + kd + ch);
      *(uint4*)(&Zl[r * LDW + ch]) = l4;
    }
    // --- stage W with fused fp32 -> bf16 hi/lo split ----------------------
    for (int idx = t; idx < 1024; idx += 256) {
      const int r  = idx >> 4;            // 0..63 (O row)
      const int kk = (idx & 15) << 2;     // float offset 0..60
      const float4 w4 = *(const float4*)(Wc + (size_t)r * KF + kb + kk);
      const unsigned int h0 = bf16h_rne(w4.x), h1 = bf16h_rne(w4.y);
      const unsigned int h2 = bf16h_rne(w4.z), h3 = bf16h_rne(w4.w);
      const unsigned int l0 = bf16h_rne(w4.x - bf16_to_f32(h0));
      const unsigned int l1 = bf16h_rne(w4.y - bf16_to_f32(h1));
      const unsigned int l2 = bf16h_rne(w4.z - bf16_to_f32(h2));
      const unsigned int l3 = bf16h_rne(w4.w - bf16_to_f32(h3));
      *(uint2*)(&Wh[r * LDW + (kk >> 1)]) = make_uint2(h0 | (h1 << 16), h2 | (h3 << 16));
      *(uint2*)(&Wl[r * LDW + (kk >> 1)]) = make_uint2(l0 | (l1 << 16), l2 | (l3 << 16));
    }
    __syncthreads();

    // --- two K=32 WMMA steps per stage ------------------------------------
    // A (16-bit 16x32): lane half=0 -> K 0..7 & 16..23, half=1 -> 8..15 & 24..31
    //   => two b128 at packed-dword offsets half*4 and 8+half*4.
    // B (16-bit 32x16): lane carries N; same K split as A
    //   => two b128 at packed-dword offsets half*4 and 8+half*4.
    #pragma unroll
    for (int s = 0; s < 2; ++s) {
      const int kds = s << 4;   // dword base of this K=32 step
      Frag bh[2], bl[2];
      #pragma unroll
      for (int j = 0; j < 2; ++j) {
        const unsigned int* ph = &Wh[((cp * 2 + j) * 16 + lm) * LDW + kds + half * 4];
        bh[j].q.lo = *(const uint4*)(ph);
        bh[j].q.hi = *(const uint4*)(ph + 8);
        const unsigned int* pl = &Wl[((cp * 2 + j) * 16 + lm) * LDW + kds + half * 4];
        bl[j].q.lo = *(const uint4*)(pl);
        bl[j].q.hi = *(const uint4*)(pl + 8);
      }
      #pragma unroll
      for (int r = 0; r < 2; ++r) {
        Frag ah, al;
        const unsigned int* ph = &Zh[(rh * 32 + r * 16 + lm) * LDW + kds];
        ah.q.lo = *(const uint4*)(ph + half * 4);
        ah.q.hi = *(const uint4*)(ph + 8 + half * 4);
        const unsigned int* pl = &Zl[(rh * 32 + r * 16 + lm) * LDW + kds];
        al.q.lo = *(const uint4*)(pl + half * 4);
        al.q.hi = *(const uint4*)(pl + 8 + half * 4);
        #pragma unroll
        for (int j = 0; j < 2; ++j) {
          v8f a = acc[j * 2 + r];
          a = wmma_bf16(ah.bf, bh[j].bf, a);   // hi * hi
          a = wmma_bf16(ah.bf, bl[j].bf, a);   // hi * lo
          a = wmma_bf16(al.bf, bh[j].bf, a);   // lo * hi
          acc[j * 2 + r] = a;
        }
      }
    }
    __syncthreads();
  }

  // --- epilogue: bias + sigmoid, scatter per C/D layout --------------------
  // C/D: VGPR i -> M = i (lanes 0-15) or i+8 (lanes 16-31); N = lane % 16.
  #pragma unroll
  for (int j = 0; j < 2; ++j) {
    const int col = col0 + (cp * 2 + j) * 16 + lm;
    const float bv = bias[c * OF + col];
    #pragma unroll
    for (int r = 0; r < 2; ++r) {
      #pragma unroll
      for (int i = 0; i < 8; ++i) {
        const int m    = i + 8 * half;
        const int nrow = row0 + rh * 32 + r * 16 + m;
        const float v  = acc[j * 2 + r][i] + bv;
        out[((size_t)nrow * C_ + c) * OF + col] = 1.0f / (1.0f + __expf(-v));
      }
    }
  }
}

// ---------------------------------------------------------------------------
extern "C" void kernel_launch(void* const* d_in, const int* in_sizes, int n_in,
                              void* d_out, int out_size, void* d_ws, size_t ws_size,
                              hipStream_t stream) {
  const float* x  = (const float*)d_in[0];   // (N, C, H, W) fp32
  const float* W  = (const float*)d_in[1];   // (C, OF, KF)  fp32
  const float* b  = (const float*)d_in[2];   // (C, OF)      fp32
  float* out = (float*)d_out;                // (N, C, OF)   fp32

  unsigned int* zh = (unsigned int*)d_ws;              // 32 MiB
  unsigned int* zl = zh + (size_t)N_ * C_ * KD;        // 32 MiB

  bitonic_sort_rows<<<N_ * C_, 256, 0, stream>>>(x, zh, zl);
  grouped_gemm_bf16x3<<<dim3(OF / 64, N_ / 128, C_), 256, 0, stream>>>(
      zh, zl, W, b, out);
}
